// GPT_52888227283821
// MI455X (gfx1250) — compile-verified, matrix operand order checked by reading, not run on hardware
//
#include <hip/hip_runtime.h>
#include <hip/hip_bf16.h>

// ---------------- constants ----------------
constexpr int Bsz  = 16;
constexpr int Ctot = 512;
constexpr int NHd  = 8;
constexpr int HDim = 64;
constexpr int FFd  = 2048;
constexpr int Lnum = 6;
constexpr int Ttot = 1280;           // 16*64 + 16*16
constexpr int Mrows = Bsz * Ttot;    // 20480

typedef __attribute__((ext_vector_type(16))) _Float16 v16h;
typedef __attribute__((ext_vector_type(8)))  _Float16 v8h;
typedef __attribute__((ext_vector_type(8)))  float    v8f;
typedef __attribute__((address_space(3)))    _Float16 lds_h;

// LDS row stride in halves: 32 payload + 8 pad = 80B -> bank-conflict-free
// b128 reads across 16 rows (r*20 mod 64 is a permutation).
constexpr int LSTR = 40;

// ---------------- helpers ----------------
// Load a 16x32 f16 WMMA operand slice for this lane.
// p points at (row_base + hi*8); halves 0..7 = K[hi*8..hi*8+7],
// halves 8..15 = K[16+hi*8..16+hi*8+7]  (CDNA5 16-bit A/B layout).
__device__ inline v16h load16(const _Float16* p) {
    v8h lo = *(const v8h*)(p);
    v8h hi = *(const v8h*)(p + 16);
    v16h r;
#pragma unroll
    for (int i = 0; i < 8; ++i) { r[i] = lo[i]; r[i + 8] = hi[i]; }
    return r;
}

__device__ inline v8f wmma_f16(v16h a, v16h b, v8f c) {
    return __builtin_amdgcn_wmma_f32_16x16x32_f16(
        false, a, false, b, (short)0, c, false, false);
}

// Async 16B copy: global -> LDS (GLOBAL_LOAD_ASYNC_TO_LDS_B128, ASYNCcnt)
__device__ inline void async_cp16(_Float16* lp, const _Float16* gp) {
    unsigned l = (unsigned)(unsigned long long)(lds_h*)lp;  // LDS byte offset
    asm volatile("global_load_async_to_lds_b128 %0, %1, off"
                 :: "v"(l), "v"(gp) : "memory");
}

__device__ inline void wait_async0() {
    asm volatile("s_wait_asynccnt 0x0" ::: "memory");
}

// ---------------- kernels ----------------

// x[b,t,c] = pos[t,c] + (img|rad)[b,c,...t...]
__global__ void embed_kernel(const float* __restrict__ img,
                             const float* __restrict__ rad,
                             const float* __restrict__ pos,
                             float* __restrict__ x) {
    size_t i = (size_t)blockIdx.x * 256 + threadIdx.x;
    size_t total = (size_t)Mrows * Ctot;
    if (i >= total) return;
    int c = (int)(i % Ctot);
    size_t bt = i / Ctot;
    int t = (int)(bt % Ttot);
    int b = (int)(bt / Ttot);
    float v;
    if (t < 1024) v = img[((size_t)b * Ctot + c) * 1024 + t];
    else          v = rad[((size_t)b * Ctot + c) * 256 + (t - 1024)];
    x[i] = v + pos[(size_t)t * Ctot + c];
}

// out[n*K + k] = (f16) in[k*N + n]   (transpose + convert)
__global__ void convT_kernel(const float* __restrict__ in,
                             _Float16* __restrict__ out, int K, int N) {
    size_t i = (size_t)blockIdx.x * 256 + threadIdx.x;
    if (i >= (size_t)K * N) return;
    int n = (int)(i % N);
    int k = (int)(i / N);
    out[(size_t)n * K + k] = (_Float16)in[i];
}

// LayerNorm over C=512, one wave per row, f16 output
__global__ __launch_bounds__(256) void ln_kernel(const float* __restrict__ x,
                                                 const float* __restrict__ w,
                                                 const float* __restrict__ b,
                                                 _Float16* __restrict__ out,
                                                 int rows) {
    int lane = threadIdx.x & 31, wave = threadIdx.x >> 5;
    int row = blockIdx.x * 8 + wave;
    if (row >= rows) return;
    const float* xr = x + (size_t)row * Ctot;
    float s = 0.f, s2 = 0.f;
    for (int i = lane; i < Ctot; i += 32) { float v = xr[i]; s += v; s2 += v * v; }
#pragma unroll
    for (int m = 1; m < 32; m <<= 1) { s += __shfl_xor(s, m, 32); s2 += __shfl_xor(s2, m, 32); }
    float mean = s * (1.f / Ctot);
    float var  = s2 * (1.f / Ctot) - mean * mean;
    float inv  = rsqrtf(var + 1e-5f);
    _Float16* o = out + (size_t)row * Ctot;
    for (int i = lane; i < Ctot; i += 32)
        o[i] = (_Float16)((xr[i] - mean) * inv * w[i] + b[i]);
}

// Final LayerNorm + permuted fp32 output (img block then radar block)
__global__ __launch_bounds__(256) void lnf_kernel(const float* __restrict__ x,
                                                  const float* __restrict__ w,
                                                  const float* __restrict__ b,
                                                  float* __restrict__ out) {
    int lane = threadIdx.x & 31, wave = threadIdx.x >> 5;
    int row = blockIdx.x * 8 + wave;
    if (row >= Mrows) return;
    const float* xr = x + (size_t)row * Ctot;
    float s = 0.f, s2 = 0.f;
    for (int i = lane; i < Ctot; i += 32) { float v = xr[i]; s += v; s2 += v * v; }
#pragma unroll
    for (int m = 1; m < 32; m <<= 1) { s += __shfl_xor(s, m, 32); s2 += __shfl_xor(s2, m, 32); }
    float mean = s * (1.f / Ctot);
    float var  = s2 * (1.f / Ctot) - mean * mean;
    float inv  = rsqrtf(var + 1e-5f);
    int bb = row / Ttot, t = row - bb * Ttot;
    float* dst;
    if (t < 1024) dst = out + ((size_t)bb * 1024 + t) * Ctot;
    else dst = out + (size_t)Bsz * 1024 * Ctot + ((size_t)bb * 256 + (t - 1024)) * Ctot;
    for (int i = lane; i < Ctot; i += 32)
        dst[i] = (xr[i] - mean) * inv * w[i] + b[i];
}

// ---------------- LDS-staged f16 GEMM ----------------
// 256 threads = 8 waves (4 M-waves x 2 N-waves); block tile 128x128;
// K staged in 32-wide panels via async global->LDS copies, double-buffered.
enum { EP_QK = 0, EP_V = 1, EP_RELU16 = 2, EP_RES = 3 };

// Copy a 128x32-half panel (row-major, leading dim ldk) into LDS (stride LSTR).
__device__ inline void stage_panel(const _Float16* gbase, int ldk,
                                   _Float16* lbase, int tid) {
#pragma unroll
    for (int it = 0; it < 2; ++it) {
        int s = tid + it * 256;          // 512 x 16B segments
        int row = s >> 2, c = (s & 3) * 8;
        async_cp16(lbase + row * LSTR + c, gbase + (size_t)row * ldk + c);
    }
}

__global__ __launch_bounds__(256) void gemm_f16(const _Float16* __restrict__ A,
                                                const _Float16* __restrict__ Bt,
                                                const float* __restrict__ bias,
                                                void* __restrict__ out,
                                                float* __restrict__ resid,
                                                int M, int N, int K, int mode) {
    __shared__ __align__(16) _Float16 Abuf[2][128 * LSTR];
    __shared__ __align__(16) _Float16 Bbuf[2][128 * LSTR];

    int tid = threadIdx.x;
    int lane = tid & 31, wave = tid >> 5;
    int mr = lane & 15, hi = lane >> 4;
    int wm = wave & 3, wn = wave >> 2;     // 4 M-waves x 2 N-waves
    int m0 = wm * 32, n0 = wn * 64;
    int tileM = blockIdx.y * 128;
    int tileN = blockIdx.x * 128;

    const _Float16* Ag = A  + (size_t)tileM * K;
    const _Float16* Bg = Bt + (size_t)tileN * K;

    v8f acc[2][4] = {};

    // prologue: stage k-panel 0
    stage_panel(Ag, K, &Abuf[0][0], tid);
    stage_panel(Bg, K, &Bbuf[0][0], tid);
    wait_async0();
    __syncthreads();

    int kTiles = K / 32;
    for (int kt = 0; kt < kTiles; ++kt) {
        int cur = kt & 1, nxt = cur ^ 1;
        if (kt + 1 < kTiles) {
            stage_panel(Ag + (kt + 1) * 32, K, &Abuf[nxt][0], tid);
            stage_panel(Bg + (kt + 1) * 32, K, &Bbuf[nxt][0], tid);
        }

        const _Float16* As = &Abuf[cur][0];
        const _Float16* Bs = &Bbuf[cur][0];
        v16h a0 = load16(As + (m0 + mr) * LSTR + hi * 8);
        v16h a1 = load16(As + (m0 + 16 + mr) * LSTR + hi * 8);
#pragma unroll
        for (int nt = 0; nt < 4; ++nt) {
            v16h b = load16(Bs + (n0 + nt * 16 + mr) * LSTR + hi * 8);
            acc[0][nt] = wmma_f16(a0, b, acc[0][nt]);
            acc[1][nt] = wmma_f16(a1, b, acc[1][nt]);
        }

        wait_async0();
        __syncthreads();
    }

    // epilogue
#pragma unroll
    for (int mt = 0; mt < 2; ++mt) {
#pragma unroll
        for (int r = 0; r < 8; ++r) {
            int row = tileM + m0 + mt * 16 + r + 8 * hi;
#pragma unroll
            for (int nt = 0; nt < 4; ++nt) {
                int col = tileN + n0 + nt * 16 + mr;
                float v = acc[mt][nt][r] + bias[col];
                if (mode == EP_RES) {
                    resid[(size_t)row * N + col] += v;
                } else if (mode == EP_RELU16) {
                    ((_Float16*)out)[(size_t)row * N + col] = (_Float16)fmaxf(v, 0.f);
                } else {
                    int bb = row / Ttot, t = row - bb * Ttot;
                    int hh = col >> 6, hd = col & 63;
                    if (mode == EP_QK)
                        ((_Float16*)out)[(((size_t)bb * NHd + hh) * Ttot + t) * HDim + hd] = (_Float16)v;
                    else // EP_V: transposed (B,H,HD,T)
                        ((_Float16*)out)[(((size_t)bb * NHd + hh) * HDim + hd) * Ttot + t] = (_Float16)v;
                }
            }
        }
    }
}

// Flash-style attention: 1 wave per (b,h,16-query tile). Q,K: (B,H,T,HD); Vt: (B,H,HD,T)
__global__ __launch_bounds__(128) void attn_kernel(const _Float16* __restrict__ Q,
                                                   const _Float16* __restrict__ Km,
                                                   const _Float16* __restrict__ Vt,
                                                   _Float16* __restrict__ Y,
                                                   float scale) {
    __shared__ __align__(16) _Float16 lds[4][16 * 32];
    int lane = threadIdx.x & 31, wave = threadIdx.x >> 5;
    int mr = lane & 15, hi = lane >> 4;
    int task = blockIdx.x * 4 + wave;          // B*NH*(T/16) tasks
    int qtile = task % (Ttot / 16);
    int bh = task / (Ttot / 16);
    int h = bh % NHd, b = bh / NHd;

    const _Float16* q = Q  + (size_t)bh * Ttot * HDim;
    const _Float16* k = Km + (size_t)bh * Ttot * HDim;
    const _Float16* v = Vt + (size_t)bh * HDim * Ttot;
    _Float16* pbuf = &lds[wave][0];

    const _Float16* qrow = q + (size_t)(qtile * 16 + mr) * HDim;
    v16h qa0 = load16(qrow + 0  + hi * 8);
    v16h qa1 = load16(qrow + 32 + hi * 8);

    float mold[8], lsum[8];
#pragma unroll
    for (int r = 0; r < 8; ++r) { mold[r] = -1e30f; lsum[r] = 0.f; }
    v8f o0 = {}, o1 = {}, o2 = {}, o3 = {};

    for (int j0 = 0; j0 < Ttot; j0 += 32) {
        const _Float16* kr0 = k + (size_t)(j0 + mr) * HDim;
        const _Float16* kr1 = k + (size_t)(j0 + 16 + mr) * HDim;
        v8f s0 = {}, s1 = {};
        s0 = wmma_f16(qa0, load16(kr0 + 0  + hi * 8), s0);
        s0 = wmma_f16(qa1, load16(kr0 + 32 + hi * 8), s0);
        s1 = wmma_f16(qa0, load16(kr1 + 0  + hi * 8), s1);
        s1 = wmma_f16(qa1, load16(kr1 + 32 + hi * 8), s1);

#pragma unroll
        for (int r = 0; r < 8; ++r) {
            float a0 = s0[r] * scale;
            float a1 = s1[r] * scale;
            float mx = fmaxf(a0, a1);
            mx = fmaxf(mx, __shfl_xor(mx, 1, 32));
            mx = fmaxf(mx, __shfl_xor(mx, 2, 32));
            mx = fmaxf(mx, __shfl_xor(mx, 4, 32));
            mx = fmaxf(mx, __shfl_xor(mx, 8, 32));
            float nm = fmaxf(mold[r], mx);
            float p0 = __expf(a0 - nm), p1 = __expf(a1 - nm);
            float ps = p0 + p1;
            ps += __shfl_xor(ps, 1, 32);
            ps += __shfl_xor(ps, 2, 32);
            ps += __shfl_xor(ps, 4, 32);
            ps += __shfl_xor(ps, 8, 32);
            float f = __expf(mold[r] - nm);
            lsum[r] = lsum[r] * f + ps;
            mold[r] = nm;
            o0[r] *= f; o1[r] *= f; o2[r] *= f; o3[r] *= f;
            pbuf[(r + 8 * hi) * 32 + mr]      = (_Float16)p0;
            pbuf[(r + 8 * hi) * 32 + 16 + mr] = (_Float16)p1;
        }

        v16h pa = load16(pbuf + (size_t)mr * 32 + hi * 8);

        o0 = wmma_f16(pa, load16(v + (size_t)(0  + mr) * Ttot + j0 + hi * 8), o0);
        o1 = wmma_f16(pa, load16(v + (size_t)(16 + mr) * Ttot + j0 + hi * 8), o1);
        o2 = wmma_f16(pa, load16(v + (size_t)(32 + mr) * Ttot + j0 + hi * 8), o2);
        o3 = wmma_f16(pa, load16(v + (size_t)(48 + mr) * Ttot + j0 + hi * 8), o3);
    }

#pragma unroll
    for (int r = 0; r < 8; ++r) {
        int t = qtile * 16 + r + 8 * hi;
        float inv = 1.f / lsum[r];
        size_t base = ((size_t)b * Ttot + t) * Ctot + h * HDim;
        Y[base + 0  + mr] = (_Float16)(o0[r] * inv);
        Y[base + 16 + mr] = (_Float16)(o1[r] * inv);
        Y[base + 32 + mr] = (_Float16)(o2[r] * inv);
        Y[base + 48 + mr] = (_Float16)(o3[r] * inv);
    }
}

// ---------------- host ----------------
extern "C" void kernel_launch(void* const* d_in, const int* in_sizes, int n_in,
                              void* d_out, int out_size, void* d_ws, size_t ws_size,
                              hipStream_t stream) {
    (void)in_sizes; (void)n_in; (void)out_size; (void)ws_size;
    const float* img  = (const float*)d_in[0];
    const float* rad  = (const float*)d_in[1];
    const float* pos  = (const float*)d_in[2];
    const float* ln1w = (const float*)d_in[3];
    const float* ln1b = (const float*)d_in[4];
    const float* Wq   = (const float*)d_in[5];
    const float* bq   = (const float*)d_in[6];
    const float* Wk   = (const float*)d_in[7];
    const float* bk   = (const float*)d_in[8];
    const float* Wv   = (const float*)d_in[9];
    const float* bv   = (const float*)d_in[10];
    const float* Wo   = (const float*)d_in[11];
    const float* bo   = (const float*)d_in[12];
    const float* ln2w = (const float*)d_in[13];
    const float* ln2b = (const float*)d_in[14];
    const float* W1   = (const float*)d_in[15];
    const float* b1   = (const float*)d_in[16];
    const float* W2   = (const float*)d_in[17];
    const float* b2   = (const float*)d_in[18];
    const float* lnfw = (const float*)d_in[19];
    const float* lnfb = (const float*)d_in[20];

    char* ws = (char*)d_ws;
    const size_t ACTF32 = (size_t)Mrows * Ctot * 4;
    const size_t ACT16  = (size_t)Mrows * Ctot * 2;
    const size_t FF16   = (size_t)Mrows * FFd  * 2;
    const size_t WCC    = (size_t)Ctot * Ctot * 2;
    const size_t WCF    = (size_t)Ctot * FFd  * 2;

    float*    x    = (float*)ws;
    _Float16* h16  = (_Float16*)(ws + ACTF32);
    _Float16* q16  = (_Float16*)(ws + ACTF32 + ACT16);
    _Float16* k16  = (_Float16*)(ws + ACTF32 + 2 * ACT16);
    _Float16* v16  = (_Float16*)(ws + ACTF32 + 3 * ACT16);
    _Float16* ff16 = (_Float16*)(ws + ACTF32 + 4 * ACT16);
    char* wb = ws + ACTF32 + 4 * ACT16 + FF16;
    _Float16* wq16 = (_Float16*)(wb);
    _Float16* wk16 = (_Float16*)(wb + WCC);
    _Float16* wv16 = (_Float16*)(wb + 2 * WCC);
    _Float16* wo16 = (_Float16*)(wb + 3 * WCC);
    _Float16* w116 = (_Float16*)(wb + 4 * WCC);
    _Float16* w216 = (_Float16*)(wb + 4 * WCC + WCF);

    const float scale = 0.125f; // 1/sqrt(64)

    {
        size_t total = (size_t)Mrows * Ctot;
        embed_kernel<<<dim3((unsigned)((total + 255) / 256)), 256, 0, stream>>>(img, rad, pos, x);
    }

    auto convT = [&](const float* in, _Float16* out, int K, int N) {
        int total = K * N;
        convT_kernel<<<dim3((total + 255) / 256), 256, 0, stream>>>(in, out, K, N);
    };
    auto gemm = [&](const _Float16* A, const _Float16* Bt, const float* bias,
                    void* out, float* resid, int M, int N, int K, int mode) {
        gemm_f16<<<dim3(N / 128, M / 128), 256, 0, stream>>>(A, Bt, bias, out, resid, M, N, K, mode);
    };

    for (int i = 0; i < Lnum; ++i) {
        convT(Wq + (size_t)i * Ctot * Ctot, wq16, Ctot, Ctot);
        convT(Wk + (size_t)i * Ctot * Ctot, wk16, Ctot, Ctot);
        convT(Wv + (size_t)i * Ctot * Ctot, wv16, Ctot, Ctot);
        convT(Wo + (size_t)i * Ctot * Ctot, wo16, Ctot, Ctot);
        convT(W1 + (size_t)i * Ctot * FFd,  w116, Ctot, FFd);
        convT(W2 + (size_t)i * FFd * Ctot,  w216, FFd,  Ctot);

        ln_kernel<<<dim3(Mrows / 8), 256, 0, stream>>>(x, ln1w + i * Ctot, ln1b + i * Ctot, h16, Mrows);

        gemm(h16, wq16, bq + i * Ctot, q16, nullptr, Mrows, Ctot, Ctot, EP_QK);
        gemm(h16, wk16, bk + i * Ctot, k16, nullptr, Mrows, Ctot, Ctot, EP_QK);
        gemm(h16, wv16, bv + i * Ctot, v16, nullptr, Mrows, Ctot, Ctot, EP_V);

        attn_kernel<<<dim3(Bsz * NHd * (Ttot / 16) / 4), 128, 0, stream>>>(q16, k16, v16, h16, scale);

        gemm(h16, wo16, bo + i * Ctot, nullptr, x, Mrows, Ctot, Ctot, EP_RES);

        ln_kernel<<<dim3(Mrows / 8), 256, 0, stream>>>(x, ln2w + i * Ctot, ln2b + i * Ctot, h16, Mrows);

        gemm(h16, w116, b1 + i * FFd, ff16, nullptr, Mrows, FFd, Ctot, EP_RELU16);
        gemm(ff16, w216, b2 + i * Ctot, nullptr, x, Mrows, Ctot, FFd, EP_RES);
    }

    lnf_kernel<<<dim3(Mrows / 8), 256, 0, stream>>>(x, lnfw, lnfb, (float*)d_out);
}
